// AttnBlock_47863115547073
// MI455X (gfx1250) — matrix operand order repaired and checked by reading, not compiled, not run on hardware
//
#include <hip/hip_runtime.h>
#include <hip/hip_bf16.h>

// ---------------------------------------------------------------------------
// AttnBlock: GN(512 groups) -> q/k/v 1x1 conv -> attention over H tokens with
// feature dim W*C -> o 1x1 conv -> residual.  B=16,H=64,W=64,C=512.
// All GEMMs via v_wmma_f32_16x16x32_f16 (f16 in, f32 accum), with block-level
// LDS staging so each streaming operand is read from L2/HBM exactly once.
// ---------------------------------------------------------------------------

typedef __attribute__((ext_vector_type(16))) _Float16 v16h;
typedef __attribute__((ext_vector_type(8)))  _Float16 v8h;
typedef __attribute__((ext_vector_type(4)))  _Float16 v4h;
typedef __attribute__((ext_vector_type(8)))  float    v8f;
typedef __attribute__((ext_vector_type(4)))  float    v4f;

#define WMMA_F16(a, b, c) \
    __builtin_amdgcn_wmma_f32_16x16x32_f16(false, (a), false, (b), (short)0, (c), false, false)

#define SHUF16(lo, hh) __builtin_shufflevector((lo), (hh), 0, 1, 2, 3, 4, 5, 6, 7, \
                                               8, 9, 10, 11, 12, 13, 14, 15)

static constexpr int B  = 16;
static constexpr int H  = 64;
static constexpr int W  = 64;
static constexpr int C  = 512;
static constexpr int HW = H * W;           // 4096 pixels per batch
static constexpr int M  = B * HW;          // 65536 rows of the projection GEMMs
static constexpr int D  = W * C;           // 32768 attention feature dim
static constexpr int KSPLIT = 8;           // split-K factor for QK^T

// A-tile LDS pitch: 528 halfs = 264 dwords; 264 % 64 == 8 -> ds_load_b128 of a
// column of rows is conflict-free across all 32 lanes (lane-halves interleave
// into the 4-dword bank groups).
static constexpr int APITCH = 528;

// -------------------------------------------------------------------- GN stats
__global__ void gn_stats_kernel(const float* __restrict__ x,
                                const float* __restrict__ gamma,
                                const float* __restrict__ beta,
                                float* __restrict__ gscale,
                                float* __restrict__ gshift) {
    int c = blockIdx.x * blockDim.x + threadIdx.x;   // 0..511
    int b = blockIdx.y;
    const float* p = x + (size_t)b * HW * C + c;
    float s = 0.f, s2 = 0.f;
    for (int i = 0; i < HW; ++i) {
        float v = p[(size_t)i * C];
        s += v; s2 += v * v;
    }
    float mean = s * (1.0f / HW);
    float var  = s2 * (1.0f / HW) - mean * mean;
    float rs   = rsqrtf(var + 1e-5f) * gamma[c];
    gscale[b * C + c] = rs;
    gshift[b * C + c] = beta[c] - mean * rs;
}

// ---------------------------------------------------- weight convert+transpose
// w_t[n][k] = (f16) w[k][n]  -> B-operand columns become contiguous per lane.
__global__ void cvt_w_kernel(const float* __restrict__ wq, const float* __restrict__ wk,
                             const float* __restrict__ wv, const float* __restrict__ wo,
                             _Float16* __restrict__ wqt, _Float16* __restrict__ wkt,
                             _Float16* __restrict__ wvt, _Float16* __restrict__ wot) {
    int idx = blockIdx.x * blockDim.x + threadIdx.x;  // 0 .. 512*512-1
    int k = idx & (C - 1);
    int n = idx >> 9;
    size_t src = (size_t)k * C + n;
    wqt[idx] = (_Float16)wq[src];
    wkt[idx] = (_Float16)wk[src];
    wvt[idx] = (_Float16)wv[src];
    wot[idx] = (_Float16)wo[src];
}

// ------------------------------------------------------- fused GN + QKV GEMMs
// Block = 16 pixels x full N=512 x {q,k,v}.  Normalized A staged once in LDS
// (x read exactly once); each of 8 waves owns 4 N-tiles -> 12 WMMAs per A-fetch.
__global__ __launch_bounds__(256) void qkv_kernel(
        const float* __restrict__ x,
        const float* __restrict__ gscale, const float* __restrict__ gshift,
        const _Float16* __restrict__ wqt, const _Float16* __restrict__ wkt,
        const _Float16* __restrict__ wvt,
        const float* __restrict__ bq, const float* __restrict__ bk,
        const float* __restrict__ bv,
        _Float16* __restrict__ qh, _Float16* __restrict__ kh,
        _Float16* __restrict__ vh) {
    __shared__ __align__(16) _Float16 xn[16][APITCH];
    int tid = threadIdx.x;
    int m0  = blockIdx.x * 16;
    int b   = m0 >> 12;                 // 4096 pixels per batch
    const float* gsb = gscale + b * C;
    const float* ghb = gshift + b * C;

    // stage + normalize 16x512 f32 -> f16 LDS (vectorized, fully coalesced)
#pragma unroll
    for (int i = 0; i < 8; ++i) {
        int e   = (tid + i * 256) * 4;          // 0..8191
        int row = e >> 9, c = e & 511;
        v4f xv = *(const v4f*)(x + (size_t)(m0 + row) * C + c);
        v4f sv = *(const v4f*)(gsb + c);
        v4f hv = *(const v4f*)(ghb + c);
        v4h o;
#pragma unroll
        for (int t = 0; t < 4; ++t) o[t] = (_Float16)(xv[t] * sv[t] + hv[t]);
        *(v4h*)(&xn[row][c]) = o;
    }
    __syncthreads();

    int wid = tid >> 5, lane = tid & 31;
    int r = lane & 15, hi = lane >> 4;
    const _Float16* wq_l = wqt + (size_t)(wid * 64 + r) * C;
    const _Float16* wk_l = wkt + (size_t)(wid * 64 + r) * C;
    const _Float16* wv_l = wvt + (size_t)(wid * 64 + r) * C;

    v8f aq[4] = {}, ak[4] = {}, av_[4] = {};
    for (int k0 = 0; k0 < C; k0 += 32) {
        v8h lo = *(const v8h*)(&xn[r][k0 + hi * 16]);
        v8h hh = *(const v8h*)(&xn[r][k0 + hi * 16 + 8]);
        v16h a = SHUF16(lo, hh);
        int kb = k0 + hi * 16;
#pragma unroll
        for (int j = 0; j < 4; ++j) {
            size_t off = (size_t)j * 16 * C + kb;
            v16h bqm = *(const v16h*)(wq_l + off);
            v16h bkm = *(const v16h*)(wk_l + off);
            v16h bvm = *(const v16h*)(wv_l + off);
            aq[j]  = WMMA_F16(a, bqm, aq[j]);
            ak[j]  = WMMA_F16(a, bkm, ak[j]);
            av_[j] = WMMA_F16(a, bvm, av_[j]);
        }
    }
#pragma unroll
    for (int j = 0; j < 4; ++j) {
        int n = wid * 64 + j * 16 + r;
        float biasq = bq[n], biask = bk[n], biasv = bv[n];
#pragma unroll
        for (int v = 0; v < 8; ++v) {
            size_t idx = (size_t)(m0 + v + hi * 8) * C + n;
            qh[idx] = (_Float16)(aq[j][v] + biasq);
            kh[idx] = (_Float16)(ak[j][v] + biask);
            vh[idx] = (_Float16)(av_[j][v] + biasv);
        }
    }
}

// ------------------------------------------------------------- QK^T (split-K)
// One wave computes a 16x64 strip (all 4 tj tiles): q rows fetched once per
// (ti, split), 4 WMMAs per A-fetch.  Deterministic split-K partial buffers.
__global__ void qk_kernel(const _Float16* __restrict__ qh,
                          const _Float16* __restrict__ kh,
                          float* __restrict__ Sp) {
    int wave = (blockIdx.x * blockDim.x + threadIdx.x) >> 5;  // 512 waves
    int lane = threadIdx.x & 31;
    int ti = wave & 3;
    int b  = (wave >> 2) & 15;
    int p  = wave >> 6;                  // 0..KSPLIT-1
    int r  = lane & 15, hi = lane >> 4;

    const _Float16* qrow  = qh + (size_t)(b * H + ti * 16 + r) * D;
    const _Float16* k0row = kh + (size_t)(b * H + r) * D;
    int kbeg = p * (D / KSPLIT);
    int kend = kbeg + (D / KSPLIT);

    v8f acc[4] = {};
    for (int k = kbeg; k < kend; k += 32) {
        __builtin_prefetch(qrow + k + 512, 0, 0);
        __builtin_prefetch(k0row + k + 512, 0, 0);
        v8h lo = *(const v8h*)(qrow + k + hi * 16);
        v8h hh = *(const v8h*)(qrow + k + hi * 16 + 8);
        v16h a = SHUF16(lo, hh);
#pragma unroll
        for (int tj = 0; tj < 4; ++tj) {
            v16h bm = *(const v16h*)(k0row + (size_t)tj * 16 * D + k + hi * 16);
            acc[tj] = WMMA_F16(a, bm, acc[tj]);
        }
    }
#pragma unroll
    for (int tj = 0; tj < 4; ++tj) {
        float* out = Sp + ((size_t)(p * B + b) * H + ti * 16) * H + tj * 16;
#pragma unroll
        for (int v = 0; v < 8; ++v)
            out[(size_t)(v + hi * 8) * H + r] = acc[tj][v];
    }
}

// ----------------------------------------------------------------- softmax row
__global__ void softmax_kernel(const float* __restrict__ Sp,
                               _Float16* __restrict__ attn) {
    int row  = blockIdx.x * (blockDim.x >> 5) + (threadIdx.x >> 5); // 0..B*H-1
    int lane = threadIdx.x & 31;
    int b = row >> 6, i = row & 63;
    const float scale = 0.044194173824159216f;  // 512^-0.5

    float a0 = 0.f, a1 = 0.f;
#pragma unroll
    for (int p = 0; p < KSPLIT; ++p) {
        const float* s = Sp + ((size_t)(p * B + b) * H + i) * H;
        a0 += s[lane];
        a1 += s[lane + 32];
    }
    a0 *= scale; a1 *= scale;
    float m = fmaxf(a0, a1);
#pragma unroll
    for (int off = 16; off; off >>= 1) m = fmaxf(m, __shfl_xor(m, off, 32));
    float e0 = __expf(a0 - m), e1 = __expf(a1 - m);
    float sum = e0 + e1;
#pragma unroll
    for (int off = 16; off; off >>= 1) sum += __shfl_xor(sum, off, 32);
    float inv = 1.0f / sum;
    attn[(size_t)row * H + lane]      = (_Float16)(e0 * inv);
    attn[(size_t)row * H + lane + 32] = (_Float16)(e1 * inv);
}

// ------------------------------------------------------------------ attn @ V
// One wave owns one 16-wide d-tile and ALL 4 token tiles: the LDS-transposed
// V tile (column-strided operand) is built once and reused by 4 WMMAs, so vh
// is read exactly once.
#define AV_WAVES 8
__global__ __launch_bounds__(256) void av_kernel(
        const _Float16* __restrict__ attn,
        const _Float16* __restrict__ vh,
        _Float16* __restrict__ o2) {
    __shared__ __align__(16) _Float16 lds[AV_WAVES][32][24];
    int tid  = threadIdx.x;
    int wid  = tid >> 5;
    int lane = tid & 31;
    int wave = blockIdx.x * AV_WAVES + wid;
    int dt = wave & 2047;              // 2048 D tiles
    int b  = wave >> 11;               // 16 batches
    int r  = lane & 15, hi = lane >> 4;

    v8f acc[4] = {};
    for (int k0 = 0; k0 < H; k0 += 32) {
        // stage V[k0+lane][dt*16 .. +15] into LDS (row per lane, 32B each)
        const _Float16* vrow = vh + (size_t)(b * H + k0 + lane) * D + dt * 16;
        *(v8h*)(&lds[wid][lane][0]) = *(const v8h*)(vrow);
        *(v8h*)(&lds[wid][lane][8]) = *(const v8h*)(vrow + 8);
        __syncthreads();
        v16h bm;                        // transposed column read (shared by ti)
#pragma unroll
        for (int t = 0; t < 16; ++t) bm[t] = lds[wid][hi * 16 + t][r];
#pragma unroll
        for (int ti = 0; ti < 4; ++ti) {
            const _Float16* arow = attn + (size_t)(b * H + ti * 16 + r) * H;
            v8h lo = *(const v8h*)(arow + k0 + hi * 16);
            v8h hh = *(const v8h*)(arow + k0 + hi * 16 + 8);
            v16h a = SHUF16(lo, hh);
            acc[ti] = WMMA_F16(a, bm, acc[ti]);
        }
        __syncthreads();
    }
#pragma unroll
    for (int ti = 0; ti < 4; ++ti)
#pragma unroll
        for (int v = 0; v < 8; ++v)
            o2[(size_t)(b * H + ti * 16 + v + hi * 8) * D + dt * 16 + r] =
                (_Float16)acc[ti][v];
}

// ------------------------------------------------- O projection + residual add
// Same block structure as qkv: A tile staged once in LDS, o2 read exactly once.
__global__ __launch_bounds__(256) void oproj_kernel(
        const _Float16* __restrict__ o2,
        const _Float16* __restrict__ wot,
        const float* __restrict__ bo,
        const float* __restrict__ x,
        float* __restrict__ out) {
    __shared__ __align__(16) _Float16 at_[16][APITCH];
    int tid = threadIdx.x;
    int m0  = blockIdx.x * 16;

#pragma unroll
    for (int i = 0; i < 4; ++i) {
        int e   = (tid + i * 256) * 8;          // 0..8191, 8 halfs per thread
        int row = e >> 9, c = e & 511;
        *(v8h*)(&at_[row][c]) = *(const v8h*)(o2 + (size_t)(m0 + row) * C + c);
    }
    __syncthreads();

    int wid = tid >> 5, lane = tid & 31;
    int r = lane & 15, hi = lane >> 4;
    const _Float16* wo_l = wot + (size_t)(wid * 64 + r) * C;

    v8f acc[4] = {};
    for (int k0 = 0; k0 < C; k0 += 32) {
        v8h lo = *(const v8h*)(&at_[r][k0 + hi * 16]);
        v8h hh = *(const v8h*)(&at_[r][k0 + hi * 16 + 8]);
        v16h a = SHUF16(lo, hh);
        int kb = k0 + hi * 16;
#pragma unroll
        for (int j = 0; j < 4; ++j) {
            v16h bm = *(const v16h*)(wo_l + (size_t)j * 16 * C + kb);
            acc[j] = WMMA_F16(a, bm, acc[j]);
        }
    }
#pragma unroll
    for (int j = 0; j < 4; ++j) {
        int n = wid * 64 + j * 16 + r;
        float bias = bo[n];
#pragma unroll
        for (int v = 0; v < 8; ++v) {
            size_t idx = (size_t)(m0 + v + hi * 8) * C + n;
            out[idx] = x[idx] + acc[j][v] + bias;
        }
    }
}

// ---------------------------------------------------------------------------
extern "C" void kernel_launch(void* const* d_in, const int* in_sizes, int n_in,
                              void* d_out, int out_size, void* d_ws, size_t ws_size,
                              hipStream_t stream) {
    const float* x     = (const float*)d_in[0];
    const float* gamma = (const float*)d_in[1];
    const float* beta  = (const float*)d_in[2];
    const float* wq    = (const float*)d_in[3];
    const float* bq    = (const float*)d_in[4];
    const float* wk    = (const float*)d_in[5];
    const float* bk    = (const float*)d_in[6];
    const float* wv    = (const float*)d_in[7];
    const float* bv    = (const float*)d_in[8];
    const float* wo    = (const float*)d_in[9];
    const float* bo    = (const float*)d_in[10];
    float* out = (float*)d_out;

    // ---- workspace carve (all offsets 256B-aligned) ----
    char* w = (char*)d_ws;
    float* gscale = (float*)w;  w += (size_t)B * C * 4;          // 32 KB
    float* gshift = (float*)w;  w += (size_t)B * C * 4;          // 32 KB
    _Float16* wqt = (_Float16*)w; w += (size_t)C * C * 2;        // 512 KB each
    _Float16* wkt = (_Float16*)w; w += (size_t)C * C * 2;
    _Float16* wvt = (_Float16*)w; w += (size_t)C * C * 2;
    _Float16* wot = (_Float16*)w; w += (size_t)C * C * 2;
    _Float16* qh  = (_Float16*)w; w += (size_t)M * C * 2;        // 64 MiB each
    _Float16* kh  = (_Float16*)w; w += (size_t)M * C * 2;
    _Float16* vh  = (_Float16*)w; w += (size_t)M * C * 2;
    float* Sp     = (float*)w;    w += (size_t)KSPLIT * B * H * H * 4; // 2 MiB
    _Float16* at  = (_Float16*)w; w += (size_t)B * H * H * 2;    // 128 KB
    _Float16* o2  = (_Float16*)w; w += (size_t)M * C * 2;        // 64 MiB
    (void)ws_size; (void)in_sizes; (void)n_in; (void)out_size;

    // 1) GroupNorm stats -> per-(b,c) scale/shift
    gn_stats_kernel<<<dim3(C / 256, B), 256, 0, stream>>>(x, gamma, beta, gscale, gshift);
    // 2) weights -> f16, transposed
    cvt_w_kernel<<<(C * C) / 256, 256, 0, stream>>>(wq, wk, wv, wo, wqt, wkt, wvt, wot);
    // 3) fused GN + q/k/v projections (LDS-staged A; x read once)
    qkv_kernel<<<M / 16, 256, 0, stream>>>(
        x, gscale, gshift, wqt, wkt, wvt, bq, bk, bv, qh, kh, vh);
    // 4) S = q k^T, split-K partials (512 waves, 4 tiles each)
    qk_kernel<<<(KSPLIT * B * 4) / 8, 256, 0, stream>>>(qh, kh, Sp);
    // 5) softmax over 64 logits per (b,i) row (1024 waves)
    softmax_kernel<<<(B * H) / 8, 256, 0, stream>>>(Sp, at);
    // 6) out2 = attn @ v (LDS-transposed V shared across 4 token tiles)
    av_kernel<<<(B * 2048) / AV_WAVES, 256, 0, stream>>>(at, vh, o2);
    // 7) O projection + bias + residual (LDS-staged A; o2 read once)
    oproj_kernel<<<M / 16, 256, 0, stream>>>(o2, wot, bo, x, out);
}
